// AdvancedTransformer_12670153523518
// MI455X (gfx1250) — compile-verified, matrix operand order checked by reading, not run on hardware
//
#include <hip/hip_runtime.h>
#include <stdint.h>

// ---------------------------------------------------------------------------
// MI455X (gfx1250) 3-block attention transformer.
// bf16 WMMA everywhere; global->LDS staging via GLOBAL_LOAD_ASYNC_TO_LDS_B128
// (ASYNCcnt) when the toolchain exposes the builtin, wide sync copies else.
// ---------------------------------------------------------------------------

#define SEQ     2048
#define DM      1024
#define NHEAD   16
#define HDIM    64
#define SM_SCALE 0.125f     // 1/sqrt(64)
#define BIGNEG  (-3.0e38f)

typedef __attribute__((ext_vector_type(16))) __bf16 v16bf;
typedef __attribute__((ext_vector_type(8)))  float  v8f;
typedef __attribute__((ext_vector_type(8)))  __bf16 bf16x8;
typedef __attribute__((ext_vector_type(4)))  int    i32x4;

#if defined(__has_builtin)
#if __has_builtin(__builtin_amdgcn_global_load_async_to_lds_b128)
#define HAVE_ASYNC_LDS 1
#endif
#if __has_builtin(__builtin_amdgcn_s_wait_asynccnt)
#define HAVE_ASYNC_WAIT 1
#endif
#endif

typedef __attribute__((address_space(1))) i32x4 gbl_i32x4;
typedef __attribute__((address_space(3))) i32x4 lds_i32x4;

// 16B global -> LDS copy. Async (ASYNCcnt) on CDNA5 toolchains, sync fallback.
static __device__ __forceinline__ void cp_b128(void* lds_dst, const void* gsrc) {
#ifdef HAVE_ASYNC_LDS
  __builtin_amdgcn_global_load_async_to_lds_b128(
      (gbl_i32x4*)gsrc, (lds_i32x4*)lds_dst, 0, 0);
#else
  *(uint4*)lds_dst = *(const uint4*)gsrc;
#endif
}

static __device__ __forceinline__ void cp_wait() {
#ifdef HAVE_ASYNC_LDS
#ifdef HAVE_ASYNC_WAIT
  __builtin_amdgcn_s_wait_asynccnt(0);
#else
  asm volatile("s_wait_asynccnt 0x0" ::: "memory");
#endif
#endif
}

// Fragment loader per CDNA5 16-bit A-layout (16x32 tile, two contiguous
// 8-element runs per lane). Serves A and B since tiles are staged with the
// contraction dim contiguous in LDS.
static __device__ __forceinline__ v16bf frag_ld(const __bf16* p, int row0, int ld, int k0) {
  const int lane = threadIdx.x & 31;
  const int r = lane & 15, h = lane >> 4;
  const __bf16* base = p + (row0 + r) * ld + k0 + h * 8;
  bf16x8 lo = *(const bf16x8*)(base);
  bf16x8 hi = *(const bf16x8*)(base + 16);
  v16bf out;
#pragma unroll
  for (int i = 0; i < 8; ++i) { out[i] = lo[i]; out[i + 8] = hi[i]; }
  return out;
}

static __device__ __forceinline__ v8f wmma_bf16(v16bf a, v16bf b, v8f c) {
  return __builtin_amdgcn_wmma_f32_16x16x32_bf16(false, a, false, b, (short)0, c, false, false);
}

// ---------------- one-time converts -----------------------------------------

__global__ void __launch_bounds__(256) cvt_bf16(const float* __restrict__ in,
                                                __bf16* __restrict__ out, int n) {
  int i = blockIdx.x * 256 + threadIdx.x;
  if (i < n) out[i] = (__bf16)in[i];
}

// W[k][n] (f32) -> WT[n][k] (bf16): GEMM B-tiles become K-contiguous.
__global__ void __launch_bounds__(256) wconv_t(const float* __restrict__ W,
                                               __bf16* __restrict__ WT) {
  __shared__ float tile[32][33];
  int n0 = blockIdx.x * 32, k0 = blockIdx.y * 32;
  int tx = threadIdx.x & 31, ty = threadIdx.x >> 5;
#pragma unroll
  for (int i = 0; i < 32; i += 8)
    tile[ty + i][tx] = W[(size_t)(k0 + ty + i) * DM + n0 + tx];
  __syncthreads();
#pragma unroll
  for (int i = 0; i < 32; i += 8)
    WT[(size_t)(n0 + ty + i) * DM + k0 + tx] = (__bf16)tile[tx][ty + i];
}

// ---------------- projection GEMM: C = A(bf16) x W + bias --------------------
// 4 waves, 64x64 tile, K-step 64: 8 WMMAs per barrier pair.

__global__ void __launch_bounds__(128)
gemm_bf16(const __bf16* __restrict__ A, const __bf16* __restrict__ WT,
          const float* __restrict__ bias, float* __restrict__ Cf,
          __bf16* __restrict__ Cb) {
  __shared__ alignas(16) __bf16 As[64 * 64];
  __shared__ alignas(16) __bf16 Bs[64 * 64];
  const int tid = threadIdx.x, wave = tid >> 5, lane = tid & 31;
  const int n0 = blockIdx.x * 64, m0 = blockIdx.y * 64;
  v8f acc[4] = {};
  for (int k0 = 0; k0 < DM; k0 += 64) {
    __syncthreads();
#pragma unroll
    for (int i = 0; i < 4; ++i) {
      int c = tid + 128 * i;            // 512 chunks of 16B per tile
      int r = c >> 3, o = (c & 7) * 8;  // row, bf16-elem offset
      cp_b128(&As[r * 64 + o], &A [(size_t)(m0 + r) * DM + k0 + o]);
      cp_b128(&Bs[r * 64 + o], &WT[(size_t)(n0 + r) * DM + k0 + o]);
    }
    if (k0 + 64 < DM)
      __builtin_prefetch(&A[(size_t)(m0 + (tid >> 1)) * DM + k0 + 64], 0, 1);
    cp_wait();
    __syncthreads();
    v16bf a0 = frag_ld(As, wave * 16, 64, 0);
    v16bf a1 = frag_ld(As, wave * 16, 64, 32);
#pragma unroll
    for (int t = 0; t < 4; ++t) {
      v16bf b0 = frag_ld(Bs, t * 16, 64, 0);
      v16bf b1 = frag_ld(Bs, t * 16, 64, 32);
      acc[t] = wmma_bf16(a0, b0, acc[t]);
      acc[t] = wmma_bf16(a1, b1, acc[t]);
    }
  }
  const int half = lane >> 4, col = lane & 15;
#pragma unroll
  for (int t = 0; t < 4; ++t) {
    int n = n0 + t * 16 + col;
    float bv = bias ? bias[n] : 0.f;
#pragma unroll
    for (int r = 0; r < 8; ++r) {
      int m = m0 + wave * 16 + half * 8 + r;
      float v = acc[t][r] + bv;
      if (Cf) Cf[(size_t)m * DM + n] = v;
      if (Cb) Cb[(size_t)m * DM + n] = (__bf16)v;
    }
  }
}

// ---------------- attention helpers -----------------------------------------

static __device__ __forceinline__ void kb_range(int mask_mode, int q0, int& lo, int& hi) {
  lo = 0; hi = SEQ;
  if (mask_mode == 1) { hi = q0 + 64; }
  else if (mask_mode == 2) {
    lo = (q0 - 128 > 0) ? q0 - 128 : 0;
    hi = (q0 + 192 < SEQ) ? q0 + 192 : SEQ;
  }
}

// stage a 64x64 bf16 head-tile (rows r0.., head col offset hc) into LDS
static __device__ __forceinline__ void stage_tile(__bf16* dst, const __bf16* src,
                                                  int r0, int hc, int tid) {
#pragma unroll
  for (int i = 0; i < 4; ++i) {
    int c = tid + 128 * i;
    int r = c >> 3, o = (c & 7) * 8;
    cp_b128(&dst[r * 64 + o], &src[(size_t)(r0 + r) * DM + hc + o]);
  }
}

// Pass 1: per-row softmax stats (max m, sum l) over the key axis.
// mask_mode: 0 none, 1 causal(k<=q), 2 local(|q-k|<=128).
__global__ void __launch_bounds__(128)
attn_stats(const __bf16* __restrict__ Qg, const __bf16* __restrict__ Kg,
           float* __restrict__ Mo, float* __restrict__ Lo, int mask_mode) {
  __shared__ alignas(16) __bf16 Qt[64 * 64];
  __shared__ alignas(16) __bf16 Kt[64 * 64];
  const int tid = threadIdx.x, wave = tid >> 5, lane = tid & 31;
  const int half = lane >> 4, col = lane & 15;
  const int q0 = blockIdx.x * 64;
  const int h  = blockIdx.y;
  const int hc = h * HDIM;
  stage_tile(Qt, Qg, q0, hc, tid);
  cp_wait();
  __syncthreads();
  v16bf aq0 = frag_ld(Qt, wave * 16, 64, 0);
  v16bf aq1 = frag_ld(Qt, wave * 16, 64, 32);
  float m_run[8], l_run[8];
#pragma unroll
  for (int r = 0; r < 8; ++r) { m_run[r] = BIGNEG; l_run[r] = 0.f; }
  int kb_lo, kb_hi; kb_range(mask_mode, q0, kb_lo, kb_hi);
  for (int kb = kb_lo; kb < kb_hi; kb += 64) {
    __syncthreads();
    stage_tile(Kt, Kg, kb, hc, tid);
    if (kb + 64 < kb_hi)
      __builtin_prefetch(&Kg[(size_t)(kb + 64 + (tid >> 1)) * DM + hc], 0, 1);
    cp_wait();
    __syncthreads();
    float sv[4][8];
#pragma unroll
    for (int t = 0; t < 4; ++t) {
      v16bf bk0 = frag_ld(Kt, t * 16, 64, 0);
      v16bf bk1 = frag_ld(Kt, t * 16, 64, 32);
      v8f sc = {};
      sc = wmma_bf16(aq0, bk0, sc);
      sc = wmma_bf16(aq1, bk1, sc);
#pragma unroll
      for (int r = 0; r < 8; ++r) {
        int q = q0 + wave * 16 + half * 8 + r;
        int k = kb + t * 16 + col;
        float s = sc[r] * SM_SCALE;
        if (mask_mode == 1 && k > q) s = BIGNEG;
        if (mask_mode == 2 && (k - q > 128 || q - k > 128)) s = BIGNEG;
        sv[t][r] = s;
      }
    }
    float rmax[8];
#pragma unroll
    for (int r = 0; r < 8; ++r) {
      float v = sv[0][r];
      v = fmaxf(v, sv[1][r]); v = fmaxf(v, sv[2][r]); v = fmaxf(v, sv[3][r]);
      rmax[r] = v;
    }
#pragma unroll
    for (int off = 1; off < 16; off <<= 1)
#pragma unroll
      for (int r = 0; r < 8; ++r) rmax[r] = fmaxf(rmax[r], __shfl_xor(rmax[r], off));
    float newm[8], rsum[8];
#pragma unroll
    for (int r = 0; r < 8; ++r) {
      newm[r] = fmaxf(m_run[r], rmax[r]);
      rsum[r] = 0.f;
    }
#pragma unroll
    for (int t = 0; t < 4; ++t)
#pragma unroll
      for (int r = 0; r < 8; ++r) rsum[r] += __expf(sv[t][r] - newm[r]);
#pragma unroll
    for (int off = 1; off < 16; off <<= 1)
#pragma unroll
      for (int r = 0; r < 8; ++r) rsum[r] += __shfl_xor(rsum[r], off);
#pragma unroll
    for (int r = 0; r < 8; ++r) {
      l_run[r] = l_run[r] * __expf(m_run[r] - newm[r]) + rsum[r];
      m_run[r] = newm[r];
    }
  }
  if (col == 0) {
#pragma unroll
    for (int r = 0; r < 8; ++r) {
      int q = q0 + wave * 16 + half * 8 + r;
      Mo[(size_t)h * SEQ + q] = m_run[r];
      Lo[(size_t)h * SEQ + q] = l_run[r];
    }
  }
}

// Pass 2: out[q,:] (+)= sum_k w(q,k) * V[k,:], stats precomputed.
// col_norm=0: w = exp(s - M[q]) / L[q]; col_norm=1: w = exp(s - M[k]) / L[k].
__global__ void __launch_bounds__(128)
attn_apply(const __bf16* __restrict__ Qg, const __bf16* __restrict__ Kg,
           const __bf16* __restrict__ Vg,
           const float* __restrict__ Mi, const float* __restrict__ Li,
           float* __restrict__ Og, int mask_mode, int col_norm, int accumulate) {
  __shared__ alignas(16) __bf16 Qt[64 * 64];
  __shared__ alignas(16) __bf16 Kt[64 * 64];
  __shared__ alignas(16) __bf16 VtT[64 * 64];      // [d][k]
  __shared__ alignas(16) __bf16 Pt[4][16 * 64];    // per-wave probability strip
  const int tid = threadIdx.x, wave = tid >> 5, lane = tid & 31;
  const int half = lane >> 4, col = lane & 15;
  const int q0 = blockIdx.x * 64;
  const int h  = blockIdx.y;
  const int hc = h * HDIM;
  stage_tile(Qt, Qg, q0, hc, tid);
  cp_wait();
  __syncthreads();
  v16bf aq0 = frag_ld(Qt, wave * 16, 64, 0);
  v16bf aq1 = frag_ld(Qt, wave * 16, 64, 32);
  float mq[8], lqi[8];
  if (!col_norm) {
#pragma unroll
    for (int r = 0; r < 8; ++r) {
      int q = q0 + wave * 16 + half * 8 + r;
      mq[r]  = Mi[(size_t)h * SEQ + q];
      lqi[r] = 1.f / Li[(size_t)h * SEQ + q];
    }
  }
  v8f o[4] = {};
  int kb_lo, kb_hi; kb_range(mask_mode, q0, kb_lo, kb_hi);
  for (int kb = kb_lo; kb < kb_hi; kb += 64) {
    __syncthreads();
    stage_tile(Kt, Kg, kb, hc, tid);
    // V transpose-stage: batch loads first for MLP, then scatter u16 to LDS.
    uint32_t tv[16];
#pragma unroll
    for (int i = 0; i < 16; ++i) {
      int e = tid + 128 * i; int r = e >> 5, c = (e & 31) * 2;
      tv[i] = *(const uint32_t*)&Vg[(size_t)(kb + r) * DM + hc + c];
    }
#pragma unroll
    for (int i = 0; i < 16; ++i) {
      int e = tid + 128 * i; int r = e >> 5, c = (e & 31) * 2;
      union { uint32_t u; __bf16 hh[2]; } cv; cv.u = tv[i];
      VtT[(c    ) * 64 + r] = cv.hh[0];
      VtT[(c + 1) * 64 + r] = cv.hh[1];
    }
    if (kb + 64 < kb_hi)
      __builtin_prefetch(&Kg[(size_t)(kb + 64 + (tid >> 1)) * DM + hc], 0, 1);
    cp_wait();
    __syncthreads();
    float mk[4], lki[4];
    if (col_norm) {
#pragma unroll
      for (int t = 0; t < 4; ++t) {
        int k = kb + t * 16 + col;
        mk[t]  = Mi[(size_t)h * SEQ + k];
        lki[t] = 1.f / Li[(size_t)h * SEQ + k];
      }
    }
#pragma unroll
    for (int t = 0; t < 4; ++t) {
      v16bf bk0 = frag_ld(Kt, t * 16, 64, 0);
      v16bf bk1 = frag_ld(Kt, t * 16, 64, 32);
      v8f sc = {};
      sc = wmma_bf16(aq0, bk0, sc);
      sc = wmma_bf16(aq1, bk1, sc);
#pragma unroll
      for (int r = 0; r < 8; ++r) {
        int q = q0 + wave * 16 + half * 8 + r;
        int k = kb + t * 16 + col;
        float s = sc[r] * SM_SCALE;
        float w;
        bool masked = (mask_mode == 1 && k > q) ||
                      (mask_mode == 2 && (k - q > 128 || q - k > 128));
        if (masked) w = 0.f;
        else if (col_norm) w = __expf(s - mk[t]) * lki[t];
        else               w = __expf(s - mq[r]) * lqi[r];
        Pt[wave][(half * 8 + r) * 64 + t * 16 + col] = (__bf16)w;
      }
    }
    __syncthreads();
    v16bf ap0 = frag_ld(&Pt[wave][0], 0, 64, 0);
    v16bf ap1 = frag_ld(&Pt[wave][0], 0, 64, 32);
#pragma unroll
    for (int nt = 0; nt < 4; ++nt) {
      v16bf bv0 = frag_ld(VtT, nt * 16, 64, 0);
      v16bf bv1 = frag_ld(VtT, nt * 16, 64, 32);
      o[nt] = wmma_bf16(ap0, bv0, o[nt]);
      o[nt] = wmma_bf16(ap1, bv1, o[nt]);
    }
  }
#pragma unroll
  for (int nt = 0; nt < 4; ++nt)
#pragma unroll
    for (int r = 0; r < 8; ++r) {
      int m = q0 + wave * 16 + half * 8 + r;
      int d = hc + nt * 16 + col;
      float v = o[nt][r];
      if (accumulate) v += Og[(size_t)m * DM + d];
      Og[(size_t)m * DM + d] = v;
    }
}

// ---------------- residual + layernorm --------------------------------------

__global__ void __launch_bounds__(256)
ln_residual(const float* __restrict__ x, const float* __restrict__ a,
            const float* __restrict__ g, const float* __restrict__ b,
            float* __restrict__ y, __bf16* __restrict__ yb) {
  __shared__ float red[2][8];
  const int row = blockIdx.x, tid = threadIdx.x;
  const int wave = tid >> 5, lane = tid & 31;
  float v[4], s = 0.f, s2 = 0.f;
#pragma unroll
  for (int i = 0; i < 4; ++i) {
    int c = tid + 256 * i;
    v[i] = x[(size_t)row * DM + c] + a[(size_t)row * DM + c];
    s += v[i]; s2 += v[i] * v[i];
  }
#pragma unroll
  for (int off = 1; off < 32; off <<= 1) {
    s  += __shfl_xor(s,  off);
    s2 += __shfl_xor(s2, off);
  }
  if (lane == 0) { red[0][wave] = s; red[1][wave] = s2; }
  __syncthreads();
  float ts = 0.f, ts2 = 0.f;
#pragma unroll
  for (int w = 0; w < 8; ++w) { ts += red[0][w]; ts2 += red[1][w]; }
  float mean = ts * (1.f / DM);
  float var  = ts2 * (1.f / DM) - mean * mean;
  float inv  = rsqrtf(var + 1e-5f);
#pragma unroll
  for (int i = 0; i < 4; ++i) {
    int c = tid + 256 * i;
    float o = g[c] * (v[i] - mean) * inv + b[c];
    y [(size_t)row * DM + c] = o;
    yb[(size_t)row * DM + c] = (__bf16)o;
  }
}

// ---------------- host orchestration ----------------------------------------

extern "C" void kernel_launch(void* const* d_in, const int* in_sizes, int n_in,
                              void* d_out, int out_size, void* d_ws, size_t ws_size,
                              hipStream_t stream) {
  (void)in_sizes; (void)n_in; (void)out_size; (void)ws_size;
  // setup_inputs() insertion order:
  // 0:x | axial 1:Wqr 2:bqr 3:Wqc 4:bqc 5:Wkr 6:bkr 7:Wkc 8:bkc 9:Wv 10:bv
  //       11:Wo 12:bo 13:ln_g 14:ln_b
  // causal 15:Wq 16:bq 17:Wk 18:bk 19:Wv 20:bv 21:Wo 22:bo 23:ln_g 24:ln_b
  // local  25..34 same layout.
  const float* xin = (const float*)d_in[0];
  auto F = [&](int i) { return (const float*)d_in[i]; };

  char* base = (char*)d_ws;
  size_t off = 0;
  auto carve = [&](size_t bytes) -> void* {
    void* p = base + off;
    off += (bytes + 255) & ~(size_t)255;
    return p;
  };

  const int widx[14] = {1, 3, 5, 7, 9, 11, 15, 17, 19, 21, 25, 27, 29, 31};
  __bf16* wT[14];
  for (int i = 0; i < 14; ++i) wT[i] = (__bf16*)carve((size_t)DM * DM * 2);
  __bf16* xb     = (__bf16*)carve((size_t)SEQ * DM * 2);
  float*  xf0    = (float*) carve((size_t)SEQ * DM * 4);
  float*  xf1    = (float*) carve((size_t)SEQ * DM * 4);
  __bf16* pq     = (__bf16*)carve((size_t)SEQ * DM * 2);
  __bf16* pk     = (__bf16*)carve((size_t)SEQ * DM * 2);
  __bf16* pq2    = (__bf16*)carve((size_t)SEQ * DM * 2);
  __bf16* pk2    = (__bf16*)carve((size_t)SEQ * DM * 2);
  __bf16* pv     = (__bf16*)carve((size_t)SEQ * DM * 2);
  float*  attnO  = (float*) carve((size_t)SEQ * DM * 4);
  __bf16* attnOb = (__bf16*)carve((size_t)SEQ * DM * 2);
  float*  oproj  = (float*) carve((size_t)SEQ * DM * 4);
  float*  m1 = (float*)carve((size_t)NHEAD * SEQ * 4);
  float*  l1 = (float*)carve((size_t)NHEAD * SEQ * 4);
  float*  m2 = (float*)carve((size_t)NHEAD * SEQ * 4);
  float*  l2 = (float*)carve((size_t)NHEAD * SEQ * 4);

  const dim3 gConv(DM / 32, DM / 32);
  for (int i = 0; i < 14; ++i)
    wconv_t<<<gConv, 256, 0, stream>>>(F(widx[i]), wT[i]);

  const int NEL = SEQ * DM;
  cvt_bf16<<<NEL / 256, 256, 0, stream>>>(xin, xb, NEL);

  const dim3 gGemm(DM / 64, SEQ / 64);
  const dim3 gAttn(SEQ / 64, NHEAD);

  // ---- axial block ----
  gemm_bf16<<<gGemm, 128, 0, stream>>>(xb, wT[0], F(2),  nullptr, pq);   // qr
  gemm_bf16<<<gGemm, 128, 0, stream>>>(xb, wT[1], F(4),  nullptr, pq2);  // qc
  gemm_bf16<<<gGemm, 128, 0, stream>>>(xb, wT[2], F(6),  nullptr, pk);   // kr
  gemm_bf16<<<gGemm, 128, 0, stream>>>(xb, wT[3], F(8),  nullptr, pk2);  // kc
  gemm_bf16<<<gGemm, 128, 0, stream>>>(xb, wT[4], F(10), nullptr, pv);   // v
  attn_stats<<<gAttn, 128, 0, stream>>>(pq, pk, m1, l1, 0);
  attn_apply<<<gAttn, 128, 0, stream>>>(pq, pk, pv, m1, l1, attnO, 0, 0, 0);
  // transposed term: scores s2[k,q] = qc[k].kc[q]; stats over q, indexed by k
  attn_stats<<<gAttn, 128, 0, stream>>>(pq2, pk2, m2, l2, 0);
  attn_apply<<<gAttn, 128, 0, stream>>>(pk2, pq2, pv, m2, l2, attnO, 0, 1, 1);
  cvt_bf16<<<NEL / 256, 256, 0, stream>>>(attnO, attnOb, NEL);
  gemm_bf16<<<gGemm, 128, 0, stream>>>(attnOb, wT[5], F(12), oproj, nullptr);
  ln_residual<<<SEQ, 256, 0, stream>>>(xin, oproj, F(13), F(14), xf0, xb);

  // ---- causal block ----
  gemm_bf16<<<gGemm, 128, 0, stream>>>(xb, wT[6], F(16), nullptr, pq);
  gemm_bf16<<<gGemm, 128, 0, stream>>>(xb, wT[7], F(18), nullptr, pk);
  gemm_bf16<<<gGemm, 128, 0, stream>>>(xb, wT[8], F(20), nullptr, pv);
  attn_stats<<<gAttn, 128, 0, stream>>>(pq, pk, m1, l1, 1);
  attn_apply<<<gAttn, 128, 0, stream>>>(pq, pk, pv, m1, l1, attnO, 1, 0, 0);
  cvt_bf16<<<NEL / 256, 256, 0, stream>>>(attnO, attnOb, NEL);
  gemm_bf16<<<gGemm, 128, 0, stream>>>(attnOb, wT[9], F(22), oproj, nullptr);
  ln_residual<<<SEQ, 256, 0, stream>>>(xf0, oproj, F(23), F(24), xf1, xb);

  // ---- local block ----
  gemm_bf16<<<gGemm, 128, 0, stream>>>(xb, wT[10], F(26), nullptr, pq);
  gemm_bf16<<<gGemm, 128, 0, stream>>>(xb, wT[11], F(28), nullptr, pk);
  gemm_bf16<<<gGemm, 128, 0, stream>>>(xb, wT[12], F(30), nullptr, pv);
  attn_stats<<<gAttn, 128, 0, stream>>>(pq, pk, m1, l1, 2);
  attn_apply<<<gAttn, 128, 0, stream>>>(pq, pk, pv, m1, l1, attnO, 2, 0, 0);
  cvt_bf16<<<NEL / 256, 256, 0, stream>>>(attnO, attnOb, NEL);
  gemm_bf16<<<gGemm, 128, 0, stream>>>(attnOb, wT[13], F(32), oproj, nullptr);
  ln_residual<<<SEQ, 256, 0, stream>>>(xf1, oproj, F(33), F(34), (float*)d_out, xb);
}
